// Block_90048284328496
// MI455X (gfx1250) — matrix operand order repaired and checked by reading, not compiled, not run on hardware
//
#include <hip/hip_runtime.h>

typedef __attribute__((ext_vector_type(16))) _Float16 v16h;
typedef __attribute__((ext_vector_type(8)))  float    v8f;

#define NB 8
#define NN 1024
#define ROWS (NB * NN)      // 8192
#define AD 256
#define VD 64
#define HH 4

#if defined(__has_builtin)
#if __has_builtin(__builtin_amdgcn_tensor_load_to_lds)
#define USE_TDM 1
#endif
#endif
#ifndef USE_TDM
#define USE_TDM 0
#endif

typedef unsigned int tdm_u32x4 __attribute__((ext_vector_type(4)));
typedef int          tdm_i32x4 __attribute__((ext_vector_type(4)));
typedef int          tdm_i32x8 __attribute__((ext_vector_type(8)));

// ---------------------------------------------------------------------------
// Tensor Data Mover: DMA a contiguous (nrows x ncols) f16 panel, row stride
// ncols elements, from global into LDS at lds_addr.  2D D#: data_size=2B,
// tensor_dim0=ncols, tensor_dim1=nrows, tile = nrows x ncols, stride0=ncols.
// Issued by one wave; caller must s_wait_tensorcnt + barrier before use.
// ---------------------------------------------------------------------------
__device__ inline void tdm_load_2d_f16(const _Float16* gsrc, unsigned lds_addr,
                                       int ncols, int nrows) {
#if USE_TDM
  unsigned long long ga = (unsigned long long)(uintptr_t)gsrc;
  tdm_u32x4 g0;
  g0.x = 1u;                                   // count=1 (valid descriptor)
  g0.y = lds_addr;                             // LDS byte address
  g0.z = (unsigned)ga;                         // global addr lo
  g0.w = (unsigned)((ga >> 32) & 0x01ffffffull) | (2u << 30);  // hi | type=2
  tdm_i32x8 g1;
  g1[0] = (int)(1u << 16);                                    // data_size=2B
  g1[1] = (int)(((unsigned)ncols & 0xffffu) << 16);           // dim0 lo16
  g1[2] = (int)(((unsigned)ncols >> 16) |
                (((unsigned)nrows & 0xffffu) << 16));         // dim0 hi | dim1 lo
  g1[3] = (int)(((unsigned)ncols & 0xffffu) << 16);           // tile_dim0
  g1[4] = (int)((unsigned)nrows & 0xffffu);                   // tile_dim1
  g1[5] = ncols;                                              // dim0 stride lo32
  g1[6] = 0;
  g1[7] = 0;
  tdm_i32x4 gz = {0, 0, 0, 0};
#if __clang_major__ >= 23
  tdm_i32x8 g4 = {0, 0, 0, 0, 0, 0, 0, 0};
  __builtin_amdgcn_tensor_load_to_lds(g0, g1, gz, gz, g4, 0);
#else
  __builtin_amdgcn_tensor_load_to_lds(g0, g1, gz, gz, 0);
#endif
#else
  (void)gsrc; (void)lds_addr; (void)ncols; (void)nrows;
#endif
}

// ---------------------------------------------------------------------------
// WMMA tile loaders for 16-bit A (16x32) / B (32x16), contiguous k order (same in A and B).
// A: lane m=l&15, kbase=(l<16)?0:16; elem j -> K=kbase+j
// B: lane n=l&15, kbase=(l<16)?0:16; elem j -> K=kbase+j
// C: VGPR r -> row = r + 8*(l>=16); col = l&15
// ---------------------------------------------------------------------------
__device__ inline v16h lds_load_a16(const _Float16* As, int lda) {
  int lane = threadIdx.x & 31;
  int m = lane & 15;
  int kb = (lane & 16) ? 16 : 0;
  v16h a;
#pragma unroll
  for (int j = 0; j < 8; ++j) a[j] = As[m * lda + kb + j];
#pragma unroll
  for (int j = 0; j < 8; ++j) a[8 + j] = As[m * lda + kb + 8 + j];
  return a;
}

__device__ inline v16h lds_load_b16(const _Float16* Bs, int ldb, int colbase) {
  int lane = threadIdx.x & 31;
  int n = colbase + (lane & 15);
  int kb = (lane & 16) ? 16 : 0;
  v16h b;
#pragma unroll
  for (int j = 0; j < 16; ++j) b[j] = Bs[(kb + j) * ldb + n];
  return b;
}

__device__ inline float leakyf(float x) { return x >= 0.f ? x : 0.2f * x; }

// f32 -> f16 weight conversion
__global__ void cvt_f16(const float* __restrict__ s, _Float16* __restrict__ d, int n) {
  int i = blockIdx.x * 256 + threadIdx.x;
  if (i < n) d[i] = (_Float16)s[i];
}

// ---------------------------------------------------------------------------
// Generic WMMA GEMM with a double-buffered TDM pipeline:
// out[rows x Ncols] = act(A[rows x K] @ Bw[K x Ncols] + bias + cin)
// Iteration ks: barrier -> wave0 issues DMA(ks+1) into the other LDS panel ->
// stage A tile -> s_wait_tensorcnt(1) (DMA(ks) landed, DMA(ks+1) in flight) ->
// barrier -> WMMA.  blockDim = 32*(Ncols/16); grid = rows/16.
// ---------------------------------------------------------------------------
__global__ void gemm_rm(const float* __restrict__ A, const _Float16* __restrict__ Bw,
                        int K, int Ncols, const float* __restrict__ bias,
                        const float* __restrict__ cin, int leaky,
                        float* __restrict__ out) {
  const int nthr = blockDim.x;
  const int tid = threadIdx.x, wave = tid >> 5, lane = tid & 31;
  const int rowbase = blockIdx.x * 16;
  __shared__ _Float16 Ah[2][16 * 32];
  __shared__ _Float16 Bh[2][32 * 256];
  const unsigned BhAddr = (unsigned)(uintptr_t)(void*)&Bh[0][0];
  v8f acc = {};
  const int total = K >> 5;
#if USE_TDM
  if (wave == 0) tdm_load_2d_f16(Bw, BhAddr, Ncols, 32);
#endif
  for (int ks = 0; ks < total; ++ks) {
    const int cb = ks & 1, nb = cb ^ 1;
    __syncthreads();  // readers of panel nb (iteration ks-1) are done
#if USE_TDM
    if (wave == 0 && ks + 1 < total)
      tdm_load_2d_f16(Bw + (size_t)(ks + 1) * 32 * Ncols,
                      BhAddr + (unsigned)(nb * 32 * 256 * 2), Ncols, 32);
#else
    for (int i = tid; i < 32 * Ncols; i += nthr)
      Bh[cb][i] = Bw[(size_t)ks * 32 * Ncols + i];
#endif
    for (int i = tid; i < 16 * 32; i += nthr) {
      int r = i >> 5, kk = i & 31;
      Ah[cb][i] = (_Float16)A[(size_t)(rowbase + r) * K + ks * 32 + kk];
    }
#if USE_TDM
    if (ks + 1 < total) __builtin_amdgcn_s_wait_tensorcnt(1);
    else                __builtin_amdgcn_s_wait_tensorcnt(0);
#endif
    __syncthreads();
    v16h a = lds_load_a16(&Ah[cb][0], 32);
    v16h bm = lds_load_b16(&Bh[cb][0], Ncols, wave * 16);
    acc = __builtin_amdgcn_wmma_f32_16x16x32_f16(false, a, false, bm, (short)0, acc,
                                                 false, false);
  }
  int col = wave * 16 + (lane & 15);
#pragma unroll
  for (int r = 0; r < 8; ++r) {
    int row = rowbase + r + ((lane & 16) ? 8 : 0);
    float v = acc[r];
    if (bias) v += bias[col];
    if (cin) v += cin[(size_t)row * Ncols + col];
    if (leaky) v = leakyf(v);
    out[(size_t)row * Ncols + col] = v;
  }
}

// ---------------------------------------------------------------------------
// Fused 7-tap conv GEMM (K = 7*256 [+256 +256]) along the chain axis.
// Segments: t=0..6 shifted X @ W7[t], optionally E1@WE1 and E2@WE2.
// Weight panels (f16) flow through the same double-buffered TDM pipeline.
// blockDim = 512 (16 waves), grid = rows/16.
// ---------------------------------------------------------------------------
__global__ void conv_gemm(const float* __restrict__ X, const _Float16* __restrict__ W7,
                          const float* __restrict__ E1, const _Float16* __restrict__ WE1,
                          const float* __restrict__ E2, const _Float16* __restrict__ WE2,
                          const float* __restrict__ bias, const float* __restrict__ cin,
                          float* __restrict__ out) {
  const int tid = threadIdx.x, wave = tid >> 5, lane = tid & 31;
  const int rowbase = blockIdx.x * 16;
  __shared__ _Float16 Ah[2][16 * 32];
  __shared__ _Float16 Bh[2][32 * 256];
  const unsigned BhAddr = (unsigned)(uintptr_t)(void*)&Bh[0][0];
  v8f acc = {};
  const int nseg = 7 + (E1 != nullptr) + (E2 != nullptr);
  const int total = nseg * 8;  // 32-wide K chunks
  auto wptr = [&](int ks) -> const _Float16* {
    int sg = ks >> 3;
    const _Float16* W = (sg < 7) ? (W7 + (size_t)sg * 65536)
                                 : (sg == 7 ? WE1 : WE2);
    return W + (size_t)((ks & 7) << 5) * 256;
  };
#if USE_TDM
  if (wave == 0) tdm_load_2d_f16(wptr(0), BhAddr, 256, 32);
#endif
  for (int ks = 0; ks < total; ++ks) {
    const int cb = ks & 1, nb = cb ^ 1;
    const int seg = ks >> 3, kc = (ks & 7) << 5;
    __syncthreads();  // readers of panel nb (iteration ks-1) are done
#if USE_TDM
    if (wave == 0 && ks + 1 < total)
      tdm_load_2d_f16(wptr(ks + 1), BhAddr + (unsigned)(nb * 32 * 256 * 2), 256, 32);
#else
    {
      const _Float16* Wsrc = wptr(ks);
      for (int i = tid; i < 32 * 256; i += 512) Bh[cb][i] = Wsrc[i];
    }
#endif
    {
      int r = tid >> 5, kk = tid & 31;   // 512 threads -> one A element each
      int g = rowbase + r, b = g >> 10, n = g & 1023;
      float v;
      if (seg < 7) {
        int nn = n + seg - 3;
        v = (nn >= 0 && nn < 1024) ? X[(size_t)((b << 10) + nn) * 256 + kc + kk] : 0.f;
      } else if (seg == 7) {
        v = E1[(size_t)g * 256 + kc + kk];
      } else {
        v = E2[(size_t)g * 256 + kc + kk];
      }
      Ah[cb][tid] = (_Float16)v;
    }
#if USE_TDM
    if (ks + 1 < total) __builtin_amdgcn_s_wait_tensorcnt(1);
    else                __builtin_amdgcn_s_wait_tensorcnt(0);
#endif
    __syncthreads();
    v16h a = lds_load_a16(&Ah[cb][0], 32);
    v16h bm = lds_load_b16(&Bh[cb][0], 256, wave * 16);
    acc = __builtin_amdgcn_wmma_f32_16x16x32_f16(false, a, false, bm, (short)0, acc,
                                                 false, false);
  }
  int col = wave * 16 + (lane & 15);
#pragma unroll
  for (int r = 0; r < 8; ++r) {
    int row = rowbase + r + ((lane & 16) ? 8 : 0);
    float v = acc[r];
    if (bias) v += bias[col];
    if (cin) v += cin[(size_t)row * 256 + col];
    out[(size_t)row * 256 + col] = v;
  }
}

// ---------------------------------------------------------------------------
// Stage A: per-node edge/node/potential embeddings + MLP-front activations.
// block = 256 threads (8 waves; wave p handles probe point p).
// ---------------------------------------------------------------------------
__global__ void prep_kernel(const float* __restrict__ pos0, const float* __restrict__ pos1,
                            const float* __restrict__ xa, const float* __restrict__ xv,
                            const float* eeW1, const float* eeb1, const float* eeb2,
                            const float* eeWg, const float* neW1, const float* neb1,
                            const float* neb2, const float* neWg, const float* ppwref,
                            const float* ppWa, const float* ppWv, const float* peWa,
                            const float* peWv, const float* c0ab, float* __restrict__ hpre,
                            float* __restrict__ h2o, float* __restrict__ embA,
                            float* __restrict__ embV) {
  const int g = blockIdx.x;
  const int b = g >> 10, n = g & 1023;
  const int tid = threadIdx.x, wave = tid >> 5, lane = tid & 31;
  __shared__ float vecs[4][3], norms[4], dvecS[3], dnormS;
  __shared__ float ppS[8][3], US[8];
  const float* p0 = pos0 + (size_t)b * 1024 * 3;
  const float* p1 = pos1 + (size_t)b * 1024 * 3;
  if (tid < 12) {
    int e = tid / 3, d = tid % 3;
    const float* P = (e < 2) ? p0 : p1;
    float v = 0.f;
    if ((e & 1) == 0) { if (n < 1023) v = P[(n + 1) * 3 + d] - P[n * 3 + d]; }
    else              { if (n > 0)    v = P[n * 3 + d] - P[(n - 1) * 3 + d]; }
    vecs[e][d] = v;
  }
  if (tid == 12) {
    float dx = p1[n * 3] - p0[n * 3];
    float dy = p1[n * 3 + 1] - p0[n * 3 + 1];
    float dz = p1[n * 3 + 2] - p0[n * 3 + 2];
    dvecS[0] = dx; dvecS[1] = dy; dvecS[2] = dz;
    dnormS = sqrtf(dx * dx + dy * dy + dz * dz + 1e-12f);
  }
  __syncthreads();
  if (tid < 4)
    norms[tid] = sqrtf(vecs[tid][0] * vecs[tid][0] + vecs[tid][1] * vecs[tid][1] +
                       vecs[tid][2] * vecs[tid][2] + 1e-12f);
  const float* xar = xa + (size_t)g * 256;
  const float* xvr = xv + (size_t)g * 192;
  float gp = 0.f;
  for (int a = lane; a < 256; a += 32) gp += xar[a] * ppWa[a * 8 + wave];
  for (int s = 16; s; s >>= 1) gp += __shfl_xor(gp, s, 32);
  float o0 = 0.f, o1 = 0.f, o2 = 0.f;
  for (int v = lane; v < 64; v += 32) {
    float wv = ppWv[wave * 64 + v];
    o0 += xvr[v * 3 + 0] * wv; o1 += xvr[v * 3 + 1] * wv; o2 += xvr[v * 3 + 2] * wv;
  }
  for (int s = 16; s; s >>= 1) {
    o0 += __shfl_xor(o0, s, 32); o1 += __shfl_xor(o1, s, 32); o2 += __shfl_xor(o2, s, 32);
  }
  if (lane == 0) {
    float U = 0.f;
    float w0 = ppwref[wave * 2], w1 = ppwref[wave * 2 + 1];
    float offd[3] = {o0, o1, o2};
    for (int d = 0; d < 3; ++d) {
      float base = w0 * p0[n * 3 + d] + w1 * p1[n * 3 + d];
      float pv = base + offd[d] * gp;
      ppS[wave][d] = pv;
      float pv2 = pv * pv;
      U += pv2 * pv2;
    }
    US[wave] = U * (1.f / 24.f);
  }
  __syncthreads();
  {
    int c = tid;
    float hp = eeb1[c];
    for (int e = 0; e < 4; ++e) hp += norms[e] * eeW1[e * 256 + c];
    hpre[(size_t)g * 256 + c] = leakyf(hp);
    float hh = neb1[c] + dnormS * neW1[c];
    h2o[(size_t)g * 256 + c] = leakyf(hh);
    float ea = eeb2[c] + neb2[c] + c0ab[c];
    for (int p = 0; p < 8; ++p) ea += US[p] * peWa[c * 8 + p];
    embA[(size_t)g * 256 + c] = ea;
  }
  if (tid < 192) {
    int v = tid / 3, d = tid % 3;
    float ev = dnormS * neWg[v] * dvecS[d];
    for (int e = 0; e < 4; ++e) {
      float gate = 0.f;
      for (int e2 = 0; e2 < 4; ++e2) gate += norms[e2] * eeWg[e2 * 256 + v * 4 + e];
      ev += gate * vecs[e][d];
    }
    for (int p = 0; p < 8; ++p) {
      float pv = ppS[p][d];
      ev += (pv * pv * pv * (1.f / 6.f)) * peWv[v * 8 + p];
    }
    embV[(size_t)g * 192 + v * 3 + d] = ev;
  }
}

// ---------------------------------------------------------------------------
// Equivariant 7-tap conv (64ch x 3 comps) with optional additive embed.
// block = 192 threads, grid = 8192 nodes.
// ---------------------------------------------------------------------------
__global__ void vconv_kernel(const float* __restrict__ xin, const float* __restrict__ w7,
                             const float* __restrict__ emb, float* __restrict__ out) {
  const int g = blockIdx.x, b = g >> 10, n = g & 1023;
  const int tid = threadIdx.x;
  __shared__ float win[7][192];
  for (int t = 0; t < 7; ++t) {
    int nn = n + t - 3;
    win[t][tid] = (nn >= 0 && nn < 1024) ? xin[(size_t)((b << 10) + nn) * 192 + tid] : 0.f;
  }
  __syncthreads();
  int w = tid / 3, d = tid % 3;
  float s = emb ? emb[(size_t)g * 192 + tid] : 0.f;
  for (int t = 0; t < 7; ++t)
    for (int v = 0; v < 64; ++v) s += win[t][v * 3 + d] * w7[(size_t)(t * 64 + v) * 64 + w];
  out[(size_t)g * 192 + tid] = s;
}

// ---------------------------------------------------------------------------
// Stage D vector path: lv chain (3 mixes + vec_roots), vr/s, p_v,
// vector GroupNorm, y_v += z_v.  block = 64 threads.
// ---------------------------------------------------------------------------
__global__ void d1_kernel(const float* lvW1, const float* lvW2, const float* lvW3,
                          const float* avpWv, const float* avpOv, const float* gv,
                          const float* __restrict__ arbuf, float* __restrict__ sbuf,
                          float* __restrict__ yv) {
  const int g = blockIdx.x;
  const int tid = threadIdx.x;
  __shared__ float row[64][3], tmp[64][3], vr[32][3], ar[32];
  float* yr = yv + (size_t)g * 192;
  for (int d = 0; d < 3; ++d) row[tid][d] = yr[tid * 3 + d];
  if (tid < 32) ar[tid] = arbuf[(size_t)g * 32 + tid];
  __syncthreads();
  float a0 = 0, a1 = 0, a2 = 0;
  for (int v = 0; v < 64; ++v) {
    float w = lvW1[v * 64 + tid];
    a0 += row[v][0] * w; a1 += row[v][1] * w; a2 += row[v][2] * w;
  }
  float f = rsqrtf(sqrtf(1.f + a0 * a0 + a1 * a1 + a2 * a2));
  tmp[tid][0] = a0 * f; tmp[tid][1] = a1 * f; tmp[tid][2] = a2 * f;
  if (tid < 32) {
    float v0 = 0, v1 = 0, v2 = 0;
    for (int v = 0; v < 64; ++v) {
      float w = avpWv[v * 32 + tid];
      v0 += row[v][0] * w; v1 += row[v][1] * w; v2 += row[v][2] * w;
    }
    vr[tid][0] = v0; vr[tid][1] = v1; vr[tid][2] = v2;
    sbuf[(size_t)g * 32 + tid] = v0 * v0 + v1 * v1 + v2 * v2;
  }
  __syncthreads();
  a0 = a1 = a2 = 0;
  for (int v = 0; v < 64; ++v) {
    float w = lvW2[v * 64 + tid];
    a0 += tmp[v][0] * w; a1 += tmp[v][1] * w; a2 += tmp[v][2] * w;
  }
  f = rsqrtf(sqrtf(1.f + a0 * a0 + a1 * a1 + a2 * a2));
  __syncthreads();
  tmp[tid][0] = a0 * f; tmp[tid][1] = a1 * f; tmp[tid][2] = a2 * f;
  __syncthreads();
  float z0 = 0, z1 = 0, z2 = 0;
  for (int v = 0; v < 64; ++v) {
    float w = lvW3[v * 64 + tid];
    z0 += tmp[v][0] * w; z1 += tmp[v][1] * w; z2 += tmp[v][2] * w;
  }
  for (int r = 0; r < 32; ++r) {
    float c = ar[r] * avpOv[r * 64 + tid];
    z0 += c * vr[r][0]; z1 += c * vr[r][1]; z2 += c * vr[r][2];
  }
  float sq = z0 * z0 + z1 * z1 + z2 * z2;
  sq += __shfl_xor(sq, 1, 32); sq += __shfl_xor(sq, 2, 32); sq += __shfl_xor(sq, 4, 32);
  float scale = rsqrtf(sq * 0.125f + 1e-5f) * gv[tid];
  yr[tid * 3 + 0] = row[tid][0] + z0 * scale;
  yr[tid * 3 + 1] = row[tid][1] + z1 * scale;
  yr[tid * 3 + 2] = row[tid][2] + z2 * scale;
}

// ---------------------------------------------------------------------------
// Scalar GroupNorm (8 groups of 32ch == one wave per group): out = base + GN(x)
// block = 256 threads.
// ---------------------------------------------------------------------------
__global__ void sgn_add(const float* __restrict__ xin, const float* ga, const float* be,
                        const float* __restrict__ base, float* __restrict__ out) {
  const int g = blockIdx.x, tid = threadIdx.x;
  float x = xin[(size_t)g * 256 + tid];
  float mu = x;
  for (int s = 16; s; s >>= 1) mu += __shfl_xor(mu, s, 32);
  mu *= (1.f / 32.f);
  float dv = x - mu;
  float var = dv * dv;
  for (int s = 16; s; s >>= 1) var += __shfl_xor(var, s, 32);
  var *= (1.f / 32.f);
  float xn = dv * rsqrtf(var + 1e-5f);
  out[(size_t)g * 256 + tid] = base[(size_t)g * 256 + tid] + xn * ga[tid] + be[tid];
}

// vector GroupNorm add: out = base + VGN(z); block = 64.
__global__ void vgn_add(const float* __restrict__ zin, const float* gv,
                        const float* __restrict__ base, float* __restrict__ out) {
  const int g = blockIdx.x, tid = threadIdx.x;
  float z0 = zin[(size_t)g * 192 + tid * 3 + 0];
  float z1 = zin[(size_t)g * 192 + tid * 3 + 1];
  float z2 = zin[(size_t)g * 192 + tid * 3 + 2];
  float sq = z0 * z0 + z1 * z1 + z2 * z2;
  sq += __shfl_xor(sq, 1, 32); sq += __shfl_xor(sq, 2, 32); sq += __shfl_xor(sq, 4, 32);
  float f = rsqrtf(sq * 0.125f + 1e-5f) * gv[tid];
  out[(size_t)g * 192 + tid * 3 + 0] = base[(size_t)g * 192 + tid * 3 + 0] + z0 * f;
  out[(size_t)g * 192 + tid * 3 + 1] = base[(size_t)g * 192 + tid * 3 + 1] + z1 * f;
  out[(size_t)g * 192 + tid * 3 + 2] = base[(size_t)g * 192 + tid * 3 + 2] + z2 * f;
}

// Attention probe points pk,pq per node -> pkpq[rows][6]. block = 256.
__global__ void bprobes(const float* __restrict__ ya, const float* __restrict__ yv,
                        const float* __restrict__ pos0, const float* __restrict__ pos1,
                        const float* wref, const float* Wa, const float* Wv,
                        float* __restrict__ pkpq) {
  const int g = blockIdx.x;
  const int tid = threadIdx.x, wave = tid >> 5, lane = tid & 31;
  __shared__ float gate[2], off[2][3];
  if (wave < 2) {
    float s = 0.f;
    for (int a = lane; a < 256; a += 32) s += ya[(size_t)g * 256 + a] * Wa[a * 2 + wave];
    for (int t = 16; t; t >>= 1) s += __shfl_xor(s, t, 32);
    if (lane == 0) gate[wave] = s;
  } else {
    int i = wave - 2, p = i / 3, d = i % 3;
    float s = 0.f;
    for (int v = lane; v < 64; v += 32) s += yv[(size_t)g * 192 + v * 3 + d] * Wv[p * 64 + v];
    for (int t = 16; t; t >>= 1) s += __shfl_xor(s, t, 32);
    if (lane == 0) off[p][d] = s;
  }
  __syncthreads();
  if (tid < 6) {
    int p = tid / 3, d = tid % 3;
    float base = wref[p * 2 + 0] * pos0[(size_t)g * 3 + d] + wref[p * 2 + 1] * pos1[(size_t)g * 3 + d];
    pkpq[(size_t)g * 6 + p * 3 + d] = base + off[p][d] * gate[p];
  }
}

// vv = per-head mixed vector values: out[rows][h*48 + c*3+d]. block = 256.
__global__ void vvmix(const float* __restrict__ yv, const float* Wvv,
                      float* __restrict__ out) {
  const int g = blockIdx.x, tid = threadIdx.x;
  __shared__ float row[192];
  if (tid < 192) row[tid] = yv[(size_t)g * 192 + tid];
  __syncthreads();
  if (tid < 192) {
    int h = tid / 48, rem = tid % 48, c = rem / 3, d = rem % 3;
    float s = 0.f;
    for (int v = 0; v < 64; ++v) s += row[v * 3 + d] * Wvv[v * 64 + h * 16 + c];
    out[(size_t)g * 192 + tid] = s;
  }
}

// y_v += dy_pre @ pa_Ov. block = 64.
__global__ void dvmix(const float* __restrict__ dy, const float* Ov,
                      float* __restrict__ yv) {
  const int g = blockIdx.x, tid = threadIdx.x;
  __shared__ float row[192];
  for (int i = tid; i < 192; i += 64) row[i] = dy[(size_t)g * 192 + i];
  __syncthreads();
  for (int d = 0; d < 3; ++d) {
    float s = 0.f;
    for (int v = 0; v < 64; ++v) s += row[v * 3 + d] * Ov[v * 64 + tid];
    yv[(size_t)g * 192 + tid * 3 + d] += s;
  }
}

// ---------------------------------------------------------------------------
// Proximity attention: one workgroup per (b, h, 16-row tile).
// Logits strip (16x1024) lives in LDS as f16; softmax in-LDS; probs re-used
// as WMMA A-tiles for the P@V output reductions (4 va tiles + 3 vv tiles).
// block = 256 (8 waves).
// ---------------------------------------------------------------------------
__global__ void attn_kernel(const float* __restrict__ q, const float* __restrict__ k,
                            const float* __restrict__ va, const float* __restrict__ vv,
                            const float* __restrict__ pkpq, float* __restrict__ dya,
                            float* __restrict__ dyv) {
  const int it = blockIdx.x, h = blockIdx.y, b = blockIdx.z;
  const int tid = threadIdx.x, wave = tid >> 5, lane = tid & 31;
  __shared__ _Float16 Lp[16 * 1024];   // 32 KB
  __shared__ float pqs[16][3];
  __shared__ float rsum[16];
  const float inv_r2[4] = {4.f, 1.f, 0.25f, 0.0625f};
  const float ir2 = inv_r2[h];
  const int rowbase = b * NN + it * 16;
  if (tid < 48) pqs[tid / 3][tid % 3] = pkpq[(size_t)(rowbase + tid / 3) * 6 + 3 + tid % 3];
  __syncthreads();
  const float qs = 0.17677669529663687f;  // 1/sqrt(32)
  const int m = lane & 15;
  const int kbA = (lane & 16) ? 16 : 0;
  v16h aq;
#pragma unroll
  for (int j = 0; j < 8; ++j)
    aq[j] = (_Float16)(q[(size_t)(rowbase + m) * 128 + h * 32 + kbA + j] * qs);
#pragma unroll
  for (int j = 0; j < 8; ++j)
    aq[8 + j] = (_Float16)(q[(size_t)(rowbase + m) * 128 + h * 32 + kbA + 8 + j] * qs);
  const int ncol = lane & 15;
  const int kbB = (lane & 16) ? 16 : 0;
  for (int jt = wave; jt < 64; jt += 8) {
    const int jb = b * NN + jt * 16;
    v16h bk;
#pragma unroll
    for (int j = 0; j < 16; ++j)
      bk[j] = (_Float16)k[(size_t)(jb + ncol) * 128 + h * 32 + kbB + j];
    v8f acc = {};
    acc = __builtin_amdgcn_wmma_f32_16x16x32_f16(false, aq, false, bk, (short)0, acc,
                                                 false, false);
    float pk0 = pkpq[(size_t)(jb + ncol) * 6 + 0];
    float pk1 = pkpq[(size_t)(jb + ncol) * 6 + 1];
    float pk2 = pkpq[(size_t)(jb + ncol) * 6 + 2];
#pragma unroll
    for (int r = 0; r < 8; ++r) {
      int row = r + ((lane & 16) ? 8 : 0);
      float dx = pqs[row][0] - pk0, dy = pqs[row][1] - pk1, dz = pqs[row][2] - pk2;
      float lg = acc[r] - (dx * dx + dy * dy + dz * dz) * ir2;
      Lp[row * 1024 + jt * 16 + ncol] = (_Float16)lg;
    }
  }
  __syncthreads();
  for (int row = wave * 2; row < wave * 2 + 2; ++row) {
    float mx = -1e30f;
    for (int j = lane; j < 1024; j += 32) mx = fmaxf(mx, (float)Lp[row * 1024 + j]);
    for (int s = 16; s; s >>= 1) mx = fmaxf(mx, __shfl_xor(mx, s, 32));
    float sm = 0.f;
    for (int j = lane; j < 1024; j += 32) {
      float e = __expf((float)Lp[row * 1024 + j] - mx);
      sm += e;
      Lp[row * 1024 + j] = (_Float16)e;
    }
    for (int s = 16; s; s >>= 1) sm += __shfl_xor(sm, s, 32);
    if (lane == 0) rsum[row] = sm;
  }
  __syncthreads();
  if (wave < 7) {
    const int ct = wave;  // 0..3: va 16-col tiles; 4..6: vv 16-col tiles
    v8f acc = {};
    for (int kt = 0; kt < 32; ++kt) {
      const _Float16* As = &Lp[0] + kt * 32;
      v16h ap;
#pragma unroll
      for (int j = 0; j < 8; ++j) ap[j] = As[m * 1024 + kbA + j];
#pragma unroll
      for (int j = 0; j < 8; ++j) ap[8 + j] = As[m * 1024 + kbA + 8 + j];
      v16h bv;
      const int jrow = b * NN + kt * 32 + kbB;
      if (ct < 4) {
#pragma unroll
        for (int j = 0; j < 16; ++j)
          bv[j] = (_Float16)va[(size_t)(jrow + j) * 256 + h * 64 + ct * 16 + ncol];
      } else {
        const int cc = (ct - 4) * 16 + ncol;
#pragma unroll
        for (int j = 0; j < 16; ++j)
          bv[j] = (_Float16)vv[(size_t)(jrow + j) * 192 + h * 48 + cc];
      }
      acc = __builtin_amdgcn_wmma_f32_16x16x32_f16(false, ap, false, bv, (short)0, acc,
                                                   false, false);
    }
#pragma unroll
    for (int r = 0; r < 8; ++r) {
      int row = r + ((lane & 16) ? 8 : 0);
      float v = acc[r] / rsum[row];
      if (ct < 4)
        dya[(size_t)(rowbase + row) * 256 + h * 64 + ct * 16 + ncol] = v;
      else
        dyv[(size_t)(rowbase + row) * 192 + h * 48 + (ct - 4) * 16 + ncol] = v;
    }
  }
}

// ---------------------------------------------------------------------------
extern "C" void kernel_launch(void* const* d_in, const int* in_sizes, int n_in,
                              void* d_out, int out_size, void* d_ws, size_t ws_size,
                              hipStream_t stream) {
  (void)in_sizes; (void)n_in; (void)out_size; (void)ws_size;
  const float* pos0  = (const float*)d_in[0];
  const float* pos1  = (const float*)d_in[1];
  const float* xa    = (const float*)d_in[2];
  const float* xv    = (const float*)d_in[3];
  const float* eeW1  = (const float*)d_in[4];
  const float* eeb1  = (const float*)d_in[5];
  const float* eeW2  = (const float*)d_in[6];
  const float* eeb2  = (const float*)d_in[7];
  const float* eeWg  = (const float*)d_in[8];
  const float* neW1  = (const float*)d_in[9];
  const float* neb1  = (const float*)d_in[10];
  const float* neW2  = (const float*)d_in[11];
  const float* neb2  = (const float*)d_in[12];
  const float* neWg  = (const float*)d_in[13];
  const float* ppwref= (const float*)d_in[14];
  const float* ppWa  = (const float*)d_in[15];
  const float* ppWv  = (const float*)d_in[16];
  const float* peWa  = (const float*)d_in[17];
  const float* peWv  = (const float*)d_in[18];
  const float* c0aw  = (const float*)d_in[19];
  const float* c0ab  = (const float*)d_in[20];
  const float* c0vw  = (const float*)d_in[21];
  const float* laW1  = (const float*)d_in[22];
  const float* lab1  = (const float*)d_in[23];
  const float* laW2  = (const float*)d_in[24];
  const float* lab2  = (const float*)d_in[25];
  const float* laW3  = (const float*)d_in[26];
  const float* lab3  = (const float*)d_in[27];
  const float* lvW1  = (const float*)d_in[28];
  const float* lvW2  = (const float*)d_in[29];
  const float* lvW3  = (const float*)d_in[30];
  const float* avpWa = (const float*)d_in[31];
  const float* avpWv = (const float*)d_in[32];
  const float* avpOa = (const float*)d_in[33];
  const float* avpOv = (const float*)d_in[34];
  const float* lrga  = (const float*)d_in[35];
  const float* lrba  = (const float*)d_in[36];
  const float* lrgv  = (const float*)d_in[37];
  const float* bpwref= (const float*)d_in[38];
  const float* bpWa  = (const float*)d_in[39];
  const float* bpWv  = (const float*)d_in[40];
  const float* paWq  = (const float*)d_in[41];
  const float* paWk  = (const float*)d_in[42];
  const float* paWva = (const float*)d_in[43];
  const float* paWvv = (const float*)d_in[44];
  const float* paOa  = (const float*)d_in[45];
  const float* paOv  = (const float*)d_in[46];
  const float* c1aw  = (const float*)d_in[47];
  const float* c1ab  = (const float*)d_in[48];
  const float* c1vw  = (const float*)d_in[49];
  const float* gnga  = (const float*)d_in[50];
  const float* gnba  = (const float*)d_in[51];
  const float* gngv  = (const float*)d_in[52];

  // ---- workspace layout: f32 activations then an f16 weight arena ----
  float* ws = (float*)d_ws;
  float* ya = ws;                       ws += (size_t)ROWS * 256;
  float* yv = ws;                       ws += (size_t)ROWS * 192;
  float* t1 = ws;                       ws += (size_t)ROWS * 256;
  float* t2 = ws;                       ws += (size_t)ROWS * 256;
  float* t3 = ws;                       ws += (size_t)ROWS * 256;
  float* ev = ws;                       ws += (size_t)ROWS * 192;  // emb_v / dy_v_pre
  float* v2 = ws;                       ws += (size_t)ROWS * 192;  // vv / conv1v tmp
  float* pkq = ws;                      ws += (size_t)ROWS * 6;

  _Float16* hp = (_Float16*)ws;
  _Float16* eeW2h  = hp; hp += 65536;
  _Float16* neW2h  = hp; hp += 65536;
  _Float16* c0awh  = hp; hp += 7 * 65536;
  _Float16* laW1h  = hp; hp += 65536;
  _Float16* laW2h  = hp; hp += 65536;
  _Float16* laW3h  = hp; hp += 65536;
  _Float16* avpWah = hp; hp += 8192;
  _Float16* avpOah = hp; hp += 8192;
  _Float16* paWqh  = hp; hp += 32768;
  _Float16* paWkh  = hp; hp += 32768;
  _Float16* paWvah = hp; hp += 65536;
  _Float16* paOah  = hp; hp += 65536;
  _Float16* c1awh  = hp; hp += 7 * 65536;

  float* out = (float*)d_out;
  float* out_p0 = out;
  float* out_p1 = out + (size_t)ROWS * 3;
  float* out_xa = out + (size_t)ROWS * 6;
  float* out_xv = out_xa + (size_t)ROWS * 256;

  const int GT = ROWS / 16;  // 512 row tiles

  // ---- weight pre-conversion to f16 (WMMA-native) ----
  cvt_f16<<<65536 / 256, 256, 0, stream>>>(eeW2, eeW2h, 65536);
  cvt_f16<<<65536 / 256, 256, 0, stream>>>(neW2, neW2h, 65536);
  cvt_f16<<<7 * 65536 / 256, 256, 0, stream>>>(c0aw, c0awh, 7 * 65536);
  cvt_f16<<<65536 / 256, 256, 0, stream>>>(laW1, laW1h, 65536);
  cvt_f16<<<65536 / 256, 256, 0, stream>>>(laW2, laW2h, 65536);
  cvt_f16<<<65536 / 256, 256, 0, stream>>>(laW3, laW3h, 65536);
  cvt_f16<<<8192 / 256, 256, 0, stream>>>(avpWa, avpWah, 8192);
  cvt_f16<<<8192 / 256, 256, 0, stream>>>(avpOa, avpOah, 8192);
  cvt_f16<<<32768 / 256, 256, 0, stream>>>(paWq, paWqh, 32768);
  cvt_f16<<<32768 / 256, 256, 0, stream>>>(paWk, paWkh, 32768);
  cvt_f16<<<65536 / 256, 256, 0, stream>>>(paWva, paWvah, 65536);
  cvt_f16<<<65536 / 256, 256, 0, stream>>>(paOa, paOah, 65536);
  cvt_f16<<<7 * 65536 / 256, 256, 0, stream>>>(c1aw, c1awh, 7 * 65536);

  // Stage A: embeddings + MLP-front activations
  prep_kernel<<<ROWS, 256, 0, stream>>>(pos0, pos1, xa, xv, eeW1, eeb1, eeb2, eeWg,
                                        neW1, neb1, neb2, neWg, ppwref, ppWa, ppWv,
                                        peWa, peWv, c0ab, t1, t2, t3, ev);
  // Stage B: y_a = conv0(x_a) + hpre@eeW2 + h2@neW2 + (emb_a + biases)
  conv_gemm<<<GT, 512, 0, stream>>>(xa, c0awh, t1, eeW2h, t2, neW2h, nullptr, t3, ya);
  // Stage C: y_v = vconv0(x_v) + emb_v
  vconv_kernel<<<ROWS, 192, 0, stream>>>(xv, c0vw, ev, yv);
  // Stage D: local residual
  gemm_rm<<<GT, 512, 0, stream>>>(ya, laW1h, 256, 256, lab1, nullptr, 1, t1);
  gemm_rm<<<GT, 512, 0, stream>>>(t1, laW2h, 256, 256, lab2, nullptr, 1, t2);
  gemm_rm<<<GT, 512, 0, stream>>>(t2, laW3h, 256, 256, lab3, nullptr, 0, t1);  // la
  gemm_rm<<<GT, 64, 0, stream>>>(ya, avpWah, 256, 32, nullptr, nullptr, 0, t3);  // ar
  d1_kernel<<<ROWS, 64, 0, stream>>>(lvW1, lvW2, lvW3, avpWv, avpOv, lrgv, t3,
                                     t3 + (size_t)ROWS * 32, yv);
  gemm_rm<<<GT, 512, 0, stream>>>(t3 + (size_t)ROWS * 32, avpOah, 32, 256, nullptr, t1,
                                  0, t2);  // la + p_a
  sgn_add<<<ROWS, 256, 0, stream>>>(t2, lrga, lrba, ya, ya);  // y_a += GN(...)
  // Stage E: proximity attention
  bprobes<<<ROWS, 256, 0, stream>>>(ya, yv, pos0, pos1, bpwref, bpWa, bpWv, pkq);
  gemm_rm<<<GT, 256, 0, stream>>>(ya, paWqh, 256, 128, nullptr, nullptr, 0, t1);  // q
  gemm_rm<<<GT, 256, 0, stream>>>(ya, paWkh, 256, 128, nullptr, nullptr, 0,
                                  t1 + (size_t)ROWS * 128);                       // k
  gemm_rm<<<GT, 512, 0, stream>>>(ya, paWvah, 256, 256, nullptr, nullptr, 0, t2); // va
  vvmix<<<ROWS, 256, 0, stream>>>(yv, paWvv, v2);                                 // vv
  attn_kernel<<<dim3(NN / 16, HH, NB), 256, 0, stream>>>(
      t1, t1 + (size_t)ROWS * 128, t2, v2, pkq, t3, ev);
  gemm_rm<<<GT, 512, 0, stream>>>(t3, paOah, 256, 256, nullptr, ya, 0, ya);  // y_a += dy_a
  dvmix<<<ROWS, 64, 0, stream>>>(ev, paOv, yv);                              // y_v += dy_v
  // Stage F: conv1 + GroupNorm, outer residual
  conv_gemm<<<GT, 512, 0, stream>>>(ya, c1awh, nullptr, nullptr, nullptr, nullptr,
                                    c1ab, nullptr, t1);
  sgn_add<<<ROWS, 256, 0, stream>>>(t1, gnga, gnba, xa, out_xa);
  vconv_kernel<<<ROWS, 192, 0, stream>>>(yv, c1vw, nullptr, v2);
  vgn_add<<<ROWS, 64, 0, stream>>>(v2, gngv, xv, out_xv);
  // pass-through positions
  hipMemcpyAsync(out_p0, pos0, (size_t)ROWS * 3 * sizeof(float),
                 hipMemcpyDeviceToDevice, stream);
  hipMemcpyAsync(out_p1, pos1, (size_t)ROWS * 3 * sizeof(float),
                 hipMemcpyDeviceToDevice, stream);
}
